// FluxHead_83915071029971
// MI455X (gfx1250) — matrix operand order repaired and checked by reading, not compiled, not run on hardware
//
#include <hip/hip_runtime.h>
#include <hip/hip_bf16.h>

// ---------------- types for WMMA fragments ----------------
typedef __attribute__((ext_vector_type(16))) _Float16 v16h;
typedef __attribute__((ext_vector_type(8)))  float    v8f;
typedef __attribute__((ext_vector_type(2)))  _Float16 h2;
typedef __attribute__((ext_vector_type(4)))  unsigned u32x4;
typedef __attribute__((ext_vector_type(8)))  int      i32x8;
typedef __attribute__((ext_vector_type(4)))  int      i32x4;

union AFrag { v16h v; h2 p[8]; unsigned u[8]; };

#define BB    4
#define TSEQ  4096
#define DM    1024
#define HD    128
#define TKP   1024          // pooled keys per batch (T/STRIDE)

#if defined(__has_builtin)
#if __has_builtin(__builtin_amdgcn_tensor_load_to_lds) && __has_builtin(__builtin_amdgcn_s_wait_tensorcnt)
#define HAVE_TDM 1
#endif
#endif

// A-fragment K-pair offset (16-bit A 16x32 layout, ISA 7.12.2)
__device__ __forceinline__ int kpairA(int v, int hi) {
    return (v < 4) ? (2 * v + 8 * hi) : (16 + 2 * (v - 4) + 8 * hi);
}
// B-fragment K-pair offset (same K order as A fragment)
__device__ __forceinline__ int kpairB(int v, int hi) {
    return (v < 4) ? (2 * v + 8 * hi) : (16 + 2 * (v - 4) + 8 * hi);
}

__device__ __forceinline__ v8f wmma_f16(const AFrag& a, const AFrag& b, v8f c) {
    return __builtin_amdgcn_wmma_f32_16x16x32_f16(false, a.v, false, b.v,
                                                  (short)0, c, false, false);
}

#ifdef HAVE_TDM
// ---- Tensor Data Mover: 2-D tile load, 8-byte elements, LDS padding ----
// ISA 8.3/8.4 descriptor packing. Dims/strides in 8-byte units.
// Toolchain here is the 6-arg builtin: (g0 u32x4, g1 i32x8, i32x4, i32x4, i32x8, cpol)
__device__ __forceinline__ void tdm_load_2d(
    unsigned lds_off, const void* gptr,
    unsigned tile_d0, unsigned tile_d1,
    unsigned tensor_d0, unsigned tensor_d1,
    unsigned long long t0_stride,
    unsigned pad_interval_code, unsigned pad_amount_code)
{
    unsigned long long ga = (unsigned long long)(size_t)gptr;
    u32x4 g0;
    g0[0] = 1u;                                            // count=1, user mode
    g0[1] = lds_off;                                       // lds_addr (bytes)
    g0[2] = (unsigned)(ga & 0xffffffffu);                  // global_addr lo
    g0[3] = (unsigned)((ga >> 32) & 0x01ffffffu)           // global_addr hi
          | (2u << 30);                                    // type=2 ("image")
    i32x8 g1;
    g1[0] = (int)((3u << 16)                               // data_size = 8B
                | (1u << 20)                               // pad_enable
                | (pad_interval_code << 22)
                | (pad_amount_code << 25));
    g1[1] = (int)((tensor_d0 & 0xffffu) << 16);            // abar=0 | dim0[15:0]
    g1[2] = (int)((tensor_d0 >> 16) | ((tensor_d1 & 0xffffu) << 16));
    g1[3] = (int)((tensor_d1 >> 16) | (tile_d0 << 16));
    g1[4] = (int)(tile_d1 & 0xffffu);                      // tile_dim1 | tile_dim2=0
    g1[5] = (int)(unsigned)(t0_stride & 0xffffffffu);
    g1[6] = (int)(unsigned)((t0_stride >> 32) & 0xffffu);  // | dim1_stride lo = 0
    g1[7] = 0;
    i32x4 z4 = {0, 0, 0, 0};
    i32x8 z8 = {0, 0, 0, 0, 0, 0, 0, 0};
    __builtin_amdgcn_tensor_load_to_lds(g0, g1, z4, z4, z8, 0);  // cpol = 0
}
__device__ __forceinline__ unsigned lds_off_of(const void* p) {
    // generic LDS address: low 32 bits are the in-allocation LDS byte offset
    return (unsigned)(size_t)p;
}
#endif

// =====================================================================
// Kernel 1: QKV projection (+ RoPE for Q, pool+RoPE for K, pool+T for V)
// grid: (128 row-tiles, 3 matrices), block: 256 threads (8 waves)
// =====================================================================
__global__ __launch_bounds__(256) void flux_proj_kernel(
    const float* __restrict__ x,  const float* __restrict__ Wq,
    const float* __restrict__ Wk, const float* __restrict__ Wv,
    _Float16* __restrict__ Qh, _Float16* __restrict__ Kh, _Float16* __restrict__ Vt)
{
    __shared__ _Float16 xs[128][36];   // 128 rows x 32 k (+4 pad)
    __shared__ _Float16 wsT[128][36];  // 128 cols x 32 k (+4 pad), transposed W

    const int tid  = threadIdx.x;
    const int wave = tid >> 5;
    const int lane = tid & 31;
    const int hi   = lane >> 4;
    const int ln   = lane & 15;
    const int rowTile = blockIdx.x;        // 0..127 (128 token rows each)
    const int which   = blockIdx.y;        // 0=Q, 1=K, 2=V
    const float* W = (which == 0) ? Wq : (which == 1) ? Wk : Wv;

    v8f acc[8];
    #pragma unroll
    for (int i = 0; i < 8; ++i)
        #pragma unroll
        for (int j = 0; j < 8; ++j) acc[i][j] = 0.0f;

    for (int kc = 0; kc < DM / 32; ++kc) {
        { // stage x tile (f32 -> f16)
            int r  = tid >> 1;
            int cb = (tid & 1) * 16;
            const float* src = x + (size_t)(rowTile * 128 + r) * DM + kc * 32 + cb;
            #pragma unroll
            for (int i = 0; i < 16; ++i) xs[r][cb + i] = (_Float16)src[i];
        }
        { // stage W chunk transposed (f32 -> f16)
            int r  = tid >> 3;
            int cb = (tid & 7) * 16;
            const float* src = W + (size_t)(kc * 32 + r) * HD + cb;
            #pragma unroll
            for (int i = 0; i < 16; ++i) wsT[cb + i][r] = (_Float16)src[i];
        }
        __syncthreads();

        AFrag a;
        #pragma unroll
        for (int v = 0; v < 8; ++v)
            a.p[v] = *(const h2*)&xs[wave * 16 + ln][kpairA(v, hi)];

        #pragma unroll
        for (int nt = 0; nt < 8; ++nt) {
            AFrag bm;
            #pragma unroll
            for (int v = 0; v < 8; ++v)
                bm.p[v] = *(const h2*)&wsT[nt * 16 + ln][kpairB(v, hi)];
            acc[nt] = wmma_f16(a, bm, acc[nt]);
        }
        __syncthreads();
    }

    // ---------------- epilogue ----------------
    const int rowBase = rowTile * 128 + wave * 16;     // flat token-row base of strip
    const float THC = -9.210340371976184f / 64.0f;     // -ln(10000)/64

    if (which == 0) {
        // RoPE(Q) then store f16. Element (token = rowBase + v + 8*hi, dim = nt*16+ln)
        #pragma unroll
        for (int nt = 0; nt < 4; ++nt) {
            int d = nt * 16 + ln;                      // < 64
            float theta = __expf((float)d * THC);
            #pragma unroll
            for (int v = 0; v < 8; ++v) {
                int g = rowBase + v + 8 * hi;
                int t = g & (TSEQ - 1);
                float ang = (float)t * theta;
                float c = __cosf(ang), s = __sinf(ang);
                float x1 = acc[nt][v], x2 = acc[nt + 4][v];
                Qh[(size_t)g * HD + d]      = (_Float16)(x1 * c - x2 * s);
                Qh[(size_t)g * HD + d + 64] = (_Float16)(x2 * c + x1 * s);
            }
        }
    } else {
        // mean-pool 4 consecutive token rows (all within this lane's 8 rows)
        float pool[8][2];
        #pragma unroll
        for (int nt = 0; nt < 8; ++nt) {
            pool[nt][0] = 0.25f * (acc[nt][0] + acc[nt][1] + acc[nt][2] + acc[nt][3]);
            pool[nt][1] = 0.25f * (acc[nt][4] + acc[nt][5] + acc[nt][6] + acc[nt][7]);
        }
        int g0  = rowBase + 8 * hi;
        int b   = g0 / TSEQ;
        int t0  = g0 & (TSEQ - 1);
        int kp0 = t0 >> 2;
        if (which == 1) {
            // RoPE(K_pool) then store f16
            #pragma unroll
            for (int nt = 0; nt < 4; ++nt) {
                int d = nt * 16 + ln;
                float theta = __expf((float)d * THC);
                #pragma unroll
                for (int j = 0; j < 2; ++j) {
                    int kp = kp0 + j;
                    float ang = (float)kp * theta;
                    float c = __cosf(ang), s = __sinf(ang);
                    float x1 = pool[nt][j], x2 = pool[nt + 4][j];
                    Kh[((size_t)b * TKP + kp) * HD + d]      = (_Float16)(x1 * c - x2 * s);
                    Kh[((size_t)b * TKP + kp) * HD + d + 64] = (_Float16)(x2 * c + x1 * s);
                }
            }
        } else {
            // V stored transposed: Vt[b][d][key]
            #pragma unroll
            for (int nt = 0; nt < 8; ++nt) {
                int d = nt * 16 + ln;
                #pragma unroll
                for (int j = 0; j < 2; ++j)
                    Vt[((size_t)b * HD + d) * TKP + kp0 + j] = (_Float16)pool[nt][j];
            }
        }
    }
}

// =====================================================================
// Kernel 2: dual-softmax attention (causal + unmasked) per 16-query tile
// grid: 512 blocks, block: 64 threads (2 waves, 1 tile/wave)
// K/V tiles streamed via TDM into padded LDS, double buffered per wave.
// =====================================================================
#define LROW 1032
#define KSTRIDE 264   // 256B K row + 8B TDM pad -> conflict-free, 8B aligned
#define VSTRIDE 72    // 64B V row + 8B TDM pad -> conflict-free, 8B aligned

__global__ __launch_bounds__(64) void flux_attn_kernel(
    const _Float16* __restrict__ Qh, const _Float16* __restrict__ Kh,
    const _Float16* __restrict__ Vt, float* __restrict__ Oc, float* __restrict__ On)
{
    __shared__ float lgt[2][16][LROW];
    __shared__ float stats[2][16][4];            // mC sC mN sN
#ifdef HAVE_TDM
    __shared__ alignas(16) char kstage[2][2][16 * KSTRIDE];   // wave x buf
    __shared__ alignas(16) char vstage[2][2][128 * VSTRIDE];  // wave x buf
#endif

    const int tid  = threadIdx.x;
    const int wv   = tid >> 5;
    const int lane = tid & 31;
    const int hi   = lane >> 4;
    const int ln   = lane & 15;
    const int tile = blockIdx.x * 2 + wv;   // 0..1023
    const int b    = tile >> 8;
    const int q0   = (tile & 255) * 16;

    // Q A-fragments for the tile: 4 K-chunks of 32 over head dim
    AFrag qa[4];
    const _Float16* qrow = Qh + (size_t)(b * TSEQ + q0 + ln) * HD;
    #pragma unroll
    for (int c2 = 0; c2 < 4; ++c2)
        #pragma unroll
        for (int v = 0; v < 8; ++v)
            qa[c2].p[v] = *(const h2*)(qrow + c2 * 32 + kpairA(v, hi));

    const float inv_scale = 0.088388347648318447f;  // 1/sqrt(128)
    const _Float16* Kbase = Kh + (size_t)b * TKP * HD;

    // -------- logits: Q @ K^T into LDS --------
#ifdef HAVE_TDM
    // K tile: 16 rows x 256B. 8B elems: tile=(32,16), row stride 32.
    // pad: 64 dwords interval (code 5), 2 dword pad (code 1) -> 264B LDS rows.
    tdm_load_2d(lds_off_of(&kstage[wv][0][0]), Kbase, 32, 16, 32, TKP, 32, 5, 1);
#endif
    for (int kt = 0; kt < TKP / 16; ++kt) {
#ifdef HAVE_TDM
        int cur = kt & 1;
        if (kt + 1 < TKP / 16) {
            tdm_load_2d(lds_off_of(&kstage[wv][cur ^ 1][0]),
                        Kbase + (size_t)(kt + 1) * 16 * HD, 32, 16, 32, TKP, 32, 5, 1);
            __builtin_amdgcn_s_wait_tensorcnt(1);   // oldest TDM (cur buf) done
        } else {
            __builtin_amdgcn_s_wait_tensorcnt(0);
        }
        const _Float16* krow = (const _Float16*)(&kstage[wv][cur][0] + (size_t)ln * KSTRIDE);
#else
        const _Float16* krow = Kbase + (size_t)(kt * 16 + ln) * HD;
#endif
        v8f c;
        #pragma unroll
        for (int j = 0; j < 8; ++j) c[j] = 0.0f;
        #pragma unroll
        for (int c2 = 0; c2 < 4; ++c2) {
            AFrag bm;
            #pragma unroll
            for (int v = 0; v < 8; ++v)
                bm.p[v] = *(const h2*)(krow + c2 * 32 + kpairB(v, hi));
            c = wmma_f16(qa[c2], bm, c);
        }
        #pragma unroll
        for (int v = 0; v < 8; ++v)
            lgt[wv][v + 8 * hi][kt * 16 + ln] = c[v] * inv_scale;
    }
    __syncthreads();

    // -------- per-row softmax stats (causal + unmasked) --------
    for (int r = 0; r < 16; ++r) {
        int kmax = (q0 + r) >> 2;        // allowed keys: k <= kmax
        float mN = -1e30f, mC = -1e30f;
        for (int k = lane; k < TKP; k += 32) {
            float t = lgt[wv][r][k];
            mN = fmaxf(mN, t);
            if (k <= kmax) mC = fmaxf(mC, t);
        }
        #pragma unroll
        for (int off = 16; off; off >>= 1) {
            mN = fmaxf(mN, __shfl_xor(mN, off, 32));
            mC = fmaxf(mC, __shfl_xor(mC, off, 32));
        }
        float sN = 0.0f, sC = 0.0f;
        for (int k = lane; k < TKP; k += 32) {
            float t = lgt[wv][r][k];
            sN += __expf(t - mN);
            if (k <= kmax) sC += __expf(t - mC);
        }
        #pragma unroll
        for (int off = 16; off; off >>= 1) {
            sN += __shfl_xor(sN, off, 32);
            sC += __shfl_xor(sC, off, 32);
        }
        if (lane == 0) {
            stats[wv][r][0] = mC; stats[wv][r][1] = sC;
            stats[wv][r][2] = mN; stats[wv][r][3] = sN;
        }
    }
    __syncthreads();

    // -------- P @ V (both variants share the V B-fragment) --------
    const float amC = stats[wv][ln][0];
    const float amN = stats[wv][ln][2];
    const int kmaxA = (q0 + ln) >> 2;
    const _Float16* Vbase = Vt + (size_t)b * HD * TKP;

    v8f accC[8], accN[8];
    #pragma unroll
    for (int i = 0; i < 8; ++i)
        #pragma unroll
        for (int j = 0; j < 8; ++j) { accC[i][j] = 0.0f; accN[i][j] = 0.0f; }

#ifdef HAVE_TDM
    // V tile for chunk kc: 128 rows (head dims) x 64B (32 keys). 8B elems:
    // tile=(8,128), memory row stride 256 (2048B). pad: 16 dwords interval
    // (code 3), 2 dword pad (code 1) -> 72B LDS rows, conflict-free.
    tdm_load_2d(lds_off_of(&vstage[wv][0][0]), Vbase, 8, 128, 256, HD, 256, 3, 1);
#endif
    for (int kc = 0; kc < TKP / 32; ++kc) {
#ifdef HAVE_TDM
        int cur = kc & 1;
        if (kc + 1 < TKP / 32) {
            tdm_load_2d(lds_off_of(&vstage[wv][cur ^ 1][0]),
                        Vbase + (size_t)(kc + 1) * 32, 8, 128, 256, HD, 256, 3, 1);
            __builtin_amdgcn_s_wait_tensorcnt(1);
        } else {
            __builtin_amdgcn_s_wait_tensorcnt(0);
        }
#endif
        AFrag pc, pn;
        #pragma unroll
        for (int v = 0; v < 8; ++v) {
            int kk = kc * 32 + kpairA(v, hi);
            float l0 = lgt[wv][ln][kk], l1 = lgt[wv][ln][kk + 1];
            float e0n = __expf(l0 - amN), e1n = __expf(l1 - amN);
            float e0c = (kk     <= kmaxA) ? __expf(l0 - amC) : 0.0f;
            float e1c = (kk + 1 <= kmaxA) ? __expf(l1 - amC) : 0.0f;
            h2 tn = {(_Float16)e0n, (_Float16)e1n};
            h2 tc = {(_Float16)e0c, (_Float16)e1c};
            pn.p[v] = tn; pc.p[v] = tc;
        }
        #pragma unroll
        for (int nt = 0; nt < 8; ++nt) {
#ifdef HAVE_TDM
            const _Float16* vrow =
                (const _Float16*)(&vstage[wv][cur][0] + (size_t)(nt * 16 + ln) * VSTRIDE);
            AFrag bm;
            #pragma unroll
            for (int v = 0; v < 8; ++v)
                bm.p[v] = *(const h2*)(vrow + kpairB(v, hi));
#else
            const _Float16* vrow = Vbase + (size_t)(nt * 16 + ln) * TKP + kc * 32;
            AFrag bm;
            #pragma unroll
            for (int v = 0; v < 8; ++v)
                bm.p[v] = *(const h2*)(vrow + kpairB(v, hi));
#endif
            accC[nt] = wmma_f16(pc, bm, accC[nt]);
            accN[nt] = wmma_f16(pn, bm, accN[nt]);
        }
    }

    // -------- epilogue: normalize and store both outputs --------
    #pragma unroll
    for (int v = 0; v < 8; ++v) {
        int r = v + 8 * hi;
        float isC = 1.0f / stats[wv][r][1];
        float isN = 1.0f / stats[wv][r][3];
        size_t base = ((size_t)b * TSEQ + q0 + r) * HD;
        #pragma unroll
        for (int nt = 0; nt < 8; ++nt) {
            Oc[base + nt * 16 + ln] = accC[nt][v] * isC;
            On[base + nt * 16 + ln] = accN[nt][v] * isN;
        }
    }
}

// =====================================================================
// Kernel 3: out[b,q] = a*Oc[b,q] + (1-a)*On[b, T-1-q],  a = sigmoid(flux_alpha)
// =====================================================================
__global__ __launch_bounds__(256) void flux_combine_kernel(
    const float* __restrict__ Oc, const float* __restrict__ On,
    const float* __restrict__ flux_alpha, float* __restrict__ out)
{
    int idx = blockIdx.x * blockDim.x + threadIdx.x;
    float a = 1.0f / (1.0f + __expf(-flux_alpha[0]));
    int d  = idx & (HD - 1);
    int q  = (idx >> 7) & (TSEQ - 1);
    int bb = idx >> 19;                       // / (TSEQ*HD)
    size_t rev = ((size_t)bb * TSEQ + (TSEQ - 1 - q)) * HD + d;
    out[idx] = a * Oc[idx] + (1.0f - a) * On[rev];
}

// =====================================================================
extern "C" void kernel_launch(void* const* d_in, const int* in_sizes, int n_in,
                              void* d_out, int out_size, void* d_ws, size_t ws_size,
                              hipStream_t stream) {
    const float* x  = (const float*)d_in[0];
    const float* Wq = (const float*)d_in[1];
    const float* Wk = (const float*)d_in[2];
    const float* Wv = (const float*)d_in[3];
    const float* flux_alpha = (const float*)d_in[9];

    // workspace layout (bytes):
    //   Qh  f16 [B*T][128]      @ 0         (4 MB)
    //   Kh  f16 [B][1024][128]  @ 4194304   (1 MB)
    //   Vt  f16 [B][128][1024]  @ 5242880   (1 MB)
    //   Oc  f32 [B][T][128]     @ 6291456   (8 MB)
    //   On  f32 [B][T][128]     @ 14680064  (8 MB)
    char* ws = (char*)d_ws;
    _Float16* Qh = (_Float16*)(ws);
    _Float16* Kh = (_Float16*)(ws + 4194304);
    _Float16* Vt = (_Float16*)(ws + 5242880);
    float*    Oc = (float*)(ws + 6291456);
    float*    On = (float*)(ws + 14680064);

    dim3 gproj(128, 3);
    flux_proj_kernel<<<gproj, 256, 0, stream>>>(x, Wq, Wk, Wv, Qh, Kh, Vt);
    flux_attn_kernel<<<512, 64, 0, stream>>>(Qh, Kh, Vt, Oc, On);
    flux_combine_kernel<<<(BB * TSEQ * HD) / 256, 256, 0, stream>>>(Oc, On, flux_alpha, (float*)d_out);
}
